// Model_54485955117333
// MI455X (gfx1250) — matrix operand order repaired and checked by reading, not compiled, not run on hardware
//
#include <hip/hip_runtime.h>
#include <hip/hip_bf16.h>
#include <math.h>

// ---------------- problem constants ----------------
static constexpr int kL    = 512;   // max length
static constexpr int kH    = 512;   // hidden
static constexpr int kB    = 32;    // batch
static constexpr int kDIN  = 832;   // GRU input dim
static constexpr int k3H   = 1536;  // 3*H gates
static constexpr int kENC  = 1024;  // 2*H bigru output
static constexpr int kCONV = 128;
static constexpr int kNBL  = kB * kL; // 16384

typedef __attribute__((ext_vector_type(16))) __bf16 v16bf;
typedef __attribute__((ext_vector_type(8)))  float  v8f;
typedef unsigned int u32x4 __attribute__((ext_vector_type(4)));
typedef int          i32x4 __attribute__((ext_vector_type(4)));
typedef int          i32x8 __attribute__((ext_vector_type(8)));

#ifndef __has_builtin
#define __has_builtin(x) 0
#endif
#if __has_builtin(__builtin_amdgcn_tensor_load_to_lds)
#define USE_TDM 1
#else
#define USE_TDM 0
#endif

// ---------------- bf16 helpers (RNE via bit math) ----------------------------
__device__ __forceinline__ __bf16 f2bf(float f) {
    union { float f; unsigned u; } in;
    in.f = f;
    unsigned u = in.u + 0x7FFFu + ((in.u >> 16) & 1u);
    union { unsigned short s; __bf16 h; } out;
    out.s = (unsigned short)(u >> 16);
    return out.h;
}

// ---------------- WMMA fragment loaders (contiguous k per lane half, same order for A and B, wave32) --------
// A tile 16x32 bf16: lane holds row m=lane&15; 16 contiguous k at
// k0+hf*16 (hf = lane>>4).
__device__ __forceinline__ v16bf frag_a_bf16(const __bf16* A, int ld, int m0, int k0, int lane) {
    const int m  = m0 + (lane & 15);
    const int hf = lane >> 4;
    const __bf16* p = A + (size_t)m * ld + k0 + hf * 8;
    v16bf f;
#pragma unroll
    for (int e = 0; e < 8; ++e) { f[e] = p[e + hf * 8]; f[8 + e] = p[8 + e + hf * 8]; }
    return f;
}
// B tile 32x16 with B[k][n] = Bm[n][k] (row-major [N,K] operand => A*B^T):
// lane holds col n=lane&15; 16 contiguous k at k0+hf*16 (32 bytes).
__device__ __forceinline__ v16bf frag_b_bf16(const __bf16* Bm, int ld, int n0, int k0, int lane) {
    const int n  = n0 + (lane & 15);
    const int hf = lane >> 4;
    const __bf16* p = Bm + (size_t)n * ld + k0 + hf * 16;
    v16bf f;
#pragma unroll
    for (int e = 0; e < 16; ++e) f[e] = p[e];
    return f;
}

__device__ __forceinline__ v8f wmma_bf16(v16bf a, v16bf b, v8f c) {
    return __builtin_amdgcn_wmma_f32_16x16x32_bf16(false, a, false, b, (short)0, c, false, false);
}

#if USE_TDM
// -------- Tensor Data Mover: 2D bf16 tile (tile_x cols contiguous, tile_y rows)
// D# bitfields per CDNA5 ISA (count=1, type=2, data_size=2B)
__device__ __forceinline__ void tdm_load_tile_2d(unsigned lds_off, unsigned long long gaddr,
                                                 int tile_x, int tile_y,
                                                 int tensor_x, int tensor_y, int stride_x) {
    u32x4 g0;
    g0[0] = 1u;                                            // count=1, user descriptor
    g0[1] = lds_off;                                       // lds_addr (bytes)
    g0[2] = (unsigned)(gaddr & 0xFFFFFFFFull);             // global_addr[31:0]
    g0[3] = (unsigned)((gaddr >> 32) & 0x1FFFFFFull) | (2u << 30);  // addr hi | type=2
    i32x8 g1;
    g1[0] = (int)(1u << 16);                               // data_size=1 (2 bytes)
    g1[1] = (int)((unsigned)(tensor_x & 0xFFFF) << 16);    // tensor_dim0[15:0]
    g1[2] = (int)(((unsigned)(tensor_x >> 16) & 0xFFFFu) |
                  ((unsigned)(tensor_y & 0xFFFF) << 16));  // tensor_dim0 hi | dim1 lo
    g1[3] = (int)(((unsigned)(tensor_y >> 16) & 0xFFFFu) |
                  ((unsigned)(tile_x & 0xFFFF) << 16));    // tensor_dim1 hi | tile_dim0
    g1[4] = (int)((unsigned)(tile_y & 0xFFFF));            // tile_dim1 (tile_dim2=0)
    g1[5] = (int)(unsigned)stride_x;                       // tensor_dim0_stride[31:0]
    g1[6] = 0;                                             // stride hi | dim1_stride lo
    g1[7] = 0;
    i32x4 g2 = {0, 0, 0, 0};
    i32x4 g3 = {0, 0, 0, 0};
#if defined(__clang_major__) && __clang_major__ >= 23
    i32x8 g4 = {0, 0, 0, 0, 0, 0, 0, 0};
    __builtin_amdgcn_tensor_load_to_lds(g0, g1, g2, g3, g4, 0);
#else
    __builtin_amdgcn_tensor_load_to_lds(g0, g1, g2, g3, 0);
#endif
}
#endif

// ---------------- f32 -> bf16 bulk convert ----------------------------------
__global__ void f32_to_bf16_kernel(const float* __restrict__ in, __bf16* __restrict__ out, int n) {
    int i = blockIdx.x * blockDim.x + threadIdx.x;
    if (i < n) out[i] = f2bf(in[i]);
}

// ---------------- embedding concat -> x bf16 [B*L, 832] ---------------------
__global__ void embed_kernel(const int* __restrict__ word, const int* __restrict__ piny,
                             const int* __restrict__ tb, const int* __restrict__ te,
                             const float* __restrict__ emb_word, const float* __restrict__ emb_piny,
                             const float* __restrict__ pe, __bf16* __restrict__ x) {
    const int bl = blockIdx.x;          // b*L + l
    const int l  = bl & (kL - 1);
    __bf16* dst = x + (size_t)bl * kDIN;
    const int w = word[bl], p = piny[bl], b0 = tb[bl], e0 = te[bl];
    for (int c = threadIdx.x; c < kDIN; c += blockDim.x) {
        float v;
        if      (c < 256) v = emb_word[(size_t)w * 256 + c];
        else if (c < 512) v = emb_piny[(size_t)p * 256 + (c - 256)];
        else if (c < 576) v = pe[l  * 64 + (c - 512)];
        else if (c < 640) v = pe[b0 * 64 + (c - 576)];
        else if (c < 704) v = pe[e0 * 64 + (c - 640)];
        else if (c < 768) v = pe[(e0 - l) * 64 + (c - 704)];
        else              v = pe[(l - b0) * 64 + (c - 768)];
        dst[c] = f2bf(v);
    }
}

// ---------------- WMMA GEMM: C[M,N] = A[M,K] * Bm[N,K]^T + bias[n] ----------
// block = 256 thr = 8 waves; tile 128(M) x 64(N); ping-pong k-loop (K % 64 == 0)
// so fragment buffers alternate roles with zero register copies.
__global__ void __launch_bounds__(256)
gemm_abt_bias_kernel(const __bf16* __restrict__ A, const __bf16* __restrict__ Bm,
                     const float* __restrict__ bias, float* __restrict__ C,
                     int M, int N, int K) {
    const int lane = threadIdx.x & 31;
    const int wave = threadIdx.x >> 5;
    const int m0   = blockIdx.y * 128 + wave * 16;
    const int n0b  = blockIdx.x * 64;
    v8f zero = {0, 0, 0, 0, 0, 0, 0, 0};
    v8f acc[4];
#pragma unroll
    for (int j = 0; j < 4; ++j) acc[j] = zero;

    v16bf a0 = frag_a_bf16(A, K, m0, 0, lane);
    v16bf b0[4];
#pragma unroll
    for (int j = 0; j < 4; ++j) b0[j] = frag_b_bf16(Bm, K, n0b + j * 16, 0, lane);

    for (int k0 = 0; k0 < K; k0 += 64) {
        // load odd set while computing even set
        v16bf a1 = frag_a_bf16(A, K, m0, k0 + 32, lane);
        v16bf b1[4];
#pragma unroll
        for (int j = 0; j < 4; ++j) b1[j] = frag_b_bf16(Bm, K, n0b + j * 16, k0 + 32, lane);
        __builtin_prefetch(A + (size_t)(m0 + (lane & 15)) * K + k0 + 128, 0, 1);
#pragma unroll
        for (int j = 0; j < 4; ++j) acc[j] = wmma_bf16(a0, b0[j], acc[j]);
        // load next even set while computing odd set
        if (k0 + 64 < K) {
            a0 = frag_a_bf16(A, K, m0, k0 + 64, lane);
#pragma unroll
            for (int j = 0; j < 4; ++j) b0[j] = frag_b_bf16(Bm, K, n0b + j * 16, k0 + 64, lane);
        }
#pragma unroll
        for (int j = 0; j < 4; ++j) acc[j] = wmma_bf16(a1, b1[j], acc[j]);
    }

    const int hf = lane >> 4, nn = lane & 15;
#pragma unroll
    for (int j = 0; j < 4; ++j) {
        const int n = n0b + j * 16 + nn;
        const float bv = bias ? bias[n] : 0.0f;
#pragma unroll
        for (int r = 0; r < 8; ++r) {
            const int m = m0 + hf * 8 + r;
            C[(size_t)m * N + n] = acc[j][r] + bv;
        }
    }
}

// ---------------- persistent GRU scan (one block per side*2+dir) ------------
// LDS: h f32 64KB + gh f32 192KB + h bf16 mirror 32KB = 288KB (<320KB/WGP)
__global__ void __launch_bounds__(1024)
gru_scan_kernel(const float* __restrict__ gi_all, const __bf16* __restrict__ whh_bf,
                const float* __restrict__ bhh_all, const int* __restrict__ len1,
                const int* __restrict__ len2, float* __restrict__ out_all) {
    const int unit = blockIdx.x;      // side*2 + dir
    const int side = unit >> 1, dir = unit & 1;
    const float*  gi  = gi_all + (size_t)unit * kNBL * k3H;
    const __bf16* Whh = whh_bf + (size_t)dir * k3H * kH;
    const float*  bhh = bhh_all + dir * k3H;
    const int*   lens = side ? len2 : len1;
    float*        out = out_all + (size_t)unit * kNBL * kH;

    extern __shared__ float smem[];
    float*  h   = smem;                                  // [32][512] f32
    float*  gh  = smem + kB * kH;                        // [32][1536] f32
    __bf16* hbf = (__bf16*)(smem + kB * kH + kB * k3H);  // [32][512] bf16 mirror

    const int tid  = threadIdx.x;
    const int lane = tid & 31, wave = tid >> 5;

    for (int i = tid; i < kB * kH; i += 1024) { h[i] = 0.0f; hbf[i] = f2bf(0.0f); }
    __syncthreads();

    for (int t = 0; t < kL; ++t) {
        // gh = h @ Whh^T + bhh : 2 Mtiles x 96 Ntiles = 192 tiles over 32 waves
#pragma unroll
        for (int i = 0; i < 6; ++i) {
            const int tile = wave * 6 + i;
            const int mt = tile & 1, nt = tile >> 1;
            const int m0 = mt * 16, n0 = nt * 16;
            v8f acc = {0, 0, 0, 0, 0, 0, 0, 0};
            // ping-pong k-loop: A from LDS bf16 (ds), B from global (L2-resident)
            v16bf a0 = frag_a_bf16(hbf, kH, m0, 0, lane);
            v16bf b0 = frag_b_bf16(Whh, kH, n0, 0, lane);
            for (int k0 = 0; k0 < kH; k0 += 64) {
                v16bf a1 = frag_a_bf16(hbf, kH, m0, k0 + 32, lane);
                v16bf b1 = frag_b_bf16(Whh, kH, n0, k0 + 32, lane);
                acc = wmma_bf16(a0, b0, acc);
                if (k0 + 64 < kH) {
                    a0 = frag_a_bf16(hbf, kH, m0, k0 + 64, lane);
                    b0 = frag_b_bf16(Whh, kH, n0, k0 + 64, lane);
                }
                acc = wmma_bf16(a1, b1, acc);
            }
            const int hf = lane >> 4, nn = lane & 15;
            const int n = n0 + nn;
#pragma unroll
            for (int r = 0; r < 8; ++r) {
                const int m = m0 + hf * 8 + r;
                gh[m * k3H + n] = acc[r] + bhh[n];
            }
        }
        __syncthreads();

        // gates: 32*512 elements / 1024 threads = 16 per thread
#pragma unroll
        for (int i = 0; i < 16; ++i) {
            const int e  = tid + i * 1024;
            const int b  = e >> 9, hh = e & 511;
            const int lenb = lens[b];
            const int tsrc = (dir == 0) ? t : ((t < lenb) ? (lenb - 1 - t) : t);
            const float* girow = gi + ((size_t)b * kL + tsrc) * k3H;
            const float ir = girow[hh], iz = girow[512 + hh], in = girow[1024 + hh];
            const float hr = gh[b * k3H + hh];
            const float hz = gh[b * k3H + 512 + hh];
            const float hn = gh[b * k3H + 1024 + hh];
            const float rg = 1.0f / (1.0f + expf(-(ir + hr)));
            const float zg = 1.0f / (1.0f + expf(-(iz + hz)));
            const float ng = tanhf(in + rg * hn);
            const float hprev = h[b * kH + hh];
            const float hnew  = (1.0f - zg) * ng + zg * hprev;
            const bool valid  = t < lenb;
            const float hkeep = valid ? hnew : hprev;
            h[b * kH + hh]   = hkeep;
            hbf[b * kH + hh] = f2bf(hkeep);
            out[((size_t)b * kL + t) * kH + hh] = valid ? hnew : 0.0f;
        }
        __syncthreads();
    }
}

// ---------------- enc = [out_f ; unreverse(out_b)] as bf16 + row norms ------
__global__ void __launch_bounds__(256)
build_enc_kernel(const float* __restrict__ out_f, const float* __restrict__ out_b,
                 const int* __restrict__ lens, __bf16* __restrict__ enc,
                 float* __restrict__ norms) {
    const int bl = blockIdx.x;
    const int b = bl >> 9, l = bl & 511;
    const int lenb = lens[b];
    const int idx = (l < lenb) ? (lenb - 1 - l) : l;
    const float* pf = out_f + (size_t)bl * kH;
    const float* pb = out_b + ((size_t)b * kL + idx) * kH;
    __bf16* pd = enc + (size_t)bl * kENC;
    float ss = 0.0f;
    for (int c = threadIdx.x; c < kENC; c += 256) {
        const float v = (c < kH) ? pf[c] : pb[c - kH];
        pd[c] = f2bf(v);
        ss += v * v;
    }
    __shared__ float red[8];
    for (int off = 16; off > 0; off >>= 1) ss += __shfl_down(ss, off, 32);
    if ((threadIdx.x & 31) == 0) red[threadIdx.x >> 5] = ss;
    __syncthreads();
    if (threadIdx.x == 0) {
        float tot = 0.0f;
#pragma unroll
        for (int i = 0; i < 8; ++i) tot += red[i];
        norms[bl] = sqrtf(tot);
    }
}

// ---------------- batched cosine sim: sim[b,i,j] via WMMA -------------------
// B panel staged through LDS by the Tensor Data Mover (double-buffered) when
// available; otherwise ping-pong global-load fragments.
__global__ void __launch_bounds__(256)
cossim_kernel(const __bf16* __restrict__ enc1, const __bf16* __restrict__ enc2,
              const float* __restrict__ n1, const float* __restrict__ n2,
              float* __restrict__ sim) {
    const int bb = blockIdx.z;
    const __bf16* A  = enc1 + (size_t)bb * kL * kENC;
    const __bf16* Bm = enc2 + (size_t)bb * kL * kENC;
    const float* nr1 = n1 + bb * kL;
    const float* nr2 = n2 + bb * kL;
    float* S = sim + (size_t)bb * kL * kL;

    const int lane = threadIdx.x & 31;
    const int wave = threadIdx.x >> 5;
    const int m0   = blockIdx.y * 128 + wave * 16;
    const int n0b  = blockIdx.x * 64;
    v8f zero = {0, 0, 0, 0, 0, 0, 0, 0};
    v8f acc[4];
#pragma unroll
    for (int j = 0; j < 4; ++j) acc[j] = zero;

#if USE_TDM
    __shared__ __bf16 ldsB[2][64 * 32];   // two 4KB chunks: 64 rows x 32 k
    const int nch = kENC / 32;
    if (wave == 0) {
        tdm_load_tile_2d((unsigned)(size_t)&ldsB[0][0],
                         (unsigned long long)(size_t)(Bm + (size_t)n0b * kENC),
                         32, 64, kENC, kL, kENC);
        __builtin_amdgcn_s_wait_tensorcnt(0);
    }
    __syncthreads();
    for (int kc = 0; kc < nch; ++kc) {
        const int cur = kc & 1;
        if (wave == 0 && kc + 1 < nch) {
            tdm_load_tile_2d((unsigned)(size_t)&ldsB[cur ^ 1][0],
                             (unsigned long long)(size_t)(Bm + (size_t)n0b * kENC + (kc + 1) * 32),
                             32, 64, kENC, kL, kENC);
        }
        v16bf a = frag_a_bf16(A, kENC, m0, kc * 32, lane);
#pragma unroll
        for (int j = 0; j < 4; ++j) {
            v16bf bf = frag_b_bf16(&ldsB[cur][0], 32, j * 16, 0, lane);
            acc[j] = wmma_bf16(a, bf, acc[j]);
        }
        if (wave == 0 && kc + 1 < nch) __builtin_amdgcn_s_wait_tensorcnt(0);
        __syncthreads();
    }
#else
    v16bf a0 = frag_a_bf16(A, kENC, m0, 0, lane);
    v16bf b0[4];
#pragma unroll
    for (int j = 0; j < 4; ++j) b0[j] = frag_b_bf16(Bm, kENC, n0b + j * 16, 0, lane);
    for (int k0 = 0; k0 < kENC; k0 += 64) {
        v16bf a1 = frag_a_bf16(A, kENC, m0, k0 + 32, lane);
        v16bf b1[4];
#pragma unroll
        for (int j = 0; j < 4; ++j) b1[j] = frag_b_bf16(Bm, kENC, n0b + j * 16, k0 + 32, lane);
#pragma unroll
        for (int j = 0; j < 4; ++j) acc[j] = wmma_bf16(a0, b0[j], acc[j]);
        if (k0 + 64 < kENC) {
            a0 = frag_a_bf16(A, kENC, m0, k0 + 64, lane);
#pragma unroll
            for (int j = 0; j < 4; ++j) b0[j] = frag_b_bf16(Bm, kENC, n0b + j * 16, k0 + 64, lane);
        }
#pragma unroll
        for (int j = 0; j < 4; ++j) acc[j] = wmma_bf16(a1, b1[j], acc[j]);
    }
#endif

    const int hf = lane >> 4, nn = lane & 15;
#pragma unroll
    for (int j = 0; j < 4; ++j) {
        const int n = n0b + j * 16 + nn;
#pragma unroll
        for (int r = 0; r < 8; ++r) {
            const int m = m0 + hf * 8 + r;
            const float den = fmaxf(nr1[m] * nr2[n], 1e-8f);
            S[(size_t)m * kL + n] = acc[j][r] / den;
        }
    }
}

// ---------------- row/col max + argmax -> max_s, rel ------------------------
__global__ void maxarg_kernel(const float* __restrict__ sim, float* __restrict__ maxs,
                              float* __restrict__ rel) {
    const int axis = blockIdx.x, b = blockIdx.y, i = threadIdx.x;
    const float* S = sim + (size_t)b * kL * kL;
    float best = -1e30f; int bi = 0;
    if (axis == 0) {
        const float* row = S + (size_t)i * kL;
        for (int j = 0; j < kL; ++j) { float v = row[j]; if (v > best) { best = v; bi = j; } }
    } else {
        for (int j = 0; j < kL; ++j) { float v = S[(size_t)j * kL + i]; if (v > best) { best = v; bi = j; } }
    }
    maxs[(axis * kB + b) * kL + i] = best;
    rel [(axis * kB + b) * kL + i] = (float)(i - bi);
}

// ---------------- conv (ks 2/3/4) + BN + ReLU + global maxpool --------------
__global__ void __launch_bounds__(128)
conv_branch_kernel(const float* __restrict__ maxs, const float* __restrict__ rel,
                   const float* __restrict__ w0, const float* __restrict__ w1,
                   const float* __restrict__ w2, const float* __restrict__ cb,
                   const float* __restrict__ gamma, const float* __restrict__ beta,
                   const float* __restrict__ mean, const float* __restrict__ var,
                   float* __restrict__ feature) {
    const int blk = blockIdx.x;       // branch*kB + b
    const int branch = blk / kB, b = blk % kB;
    __shared__ float xs[2][kL];
    for (int i = threadIdx.x; i < kL; i += 128) {
        xs[0][i] = maxs[(branch * kB + b) * kL + i];
        xs[1][i] = rel [(branch * kB + b) * kL + i];
    }
    __syncthreads();
    const int c = threadIdx.x;        // out channel
    const float* wptr[3] = {w0, w1, w2};
    const int kss[3] = {2, 3, 4};
#pragma unroll
    for (int ki = 0; ki < 3; ++ki) {
        const int ks = kss[ki];
        const float* w  = wptr[ki] + c * 2 * ks;        // [c][ic][k]
        const float bias = cb[ki * kCONV + c];
        const float mu   = mean[ki * kCONV + c];
        const float iv   = rsqrtf(var[ki * kCONV + c] + 1e-5f);
        const float ga   = gamma[ki * kCONV + c];
        const float be   = beta [ki * kCONV + c];
        float mx = -1e30f;
        for (int p = 0; p <= kL - ks; ++p) {
            float y = bias;
            for (int kk = 0; kk < ks; ++kk)
                y += xs[0][p + kk] * w[kk] + xs[1][p + kk] * w[ks + kk];
            y = (y - mu) * iv * ga + be;
            y = y > 0.0f ? y : 0.0f;
            mx = y > mx ? y : mx;
        }
        feature[b * (6 * kCONV) + branch * (3 * kCONV) + ki * kCONV + c] = mx;
    }
}

// ---------------- fc + softmax ----------------------------------------------
__global__ void fc_softmax_kernel(const float* __restrict__ feature, const float* __restrict__ fcw,
                                  const float* __restrict__ fcb, float* __restrict__ out) {
    const int b = blockIdx.x;
    const int tid = threadIdx.x;      // 64 threads = 2 waves
    const int cls = tid >> 5, lane = tid & 31;
    const float* f = feature + b * 768;
    const float* w = fcw + cls * 768;
    float s = 0.0f;
    for (int i = lane; i < 768; i += 32) s += f[i] * w[i];
    for (int off = 16; off > 0; off >>= 1) s += __shfl_down(s, off, 32);
    __shared__ float logits[2];
    if (lane == 0) logits[cls] = s + fcb[cls];
    __syncthreads();
    if (tid == 0) {
        const float a = logits[0], c = logits[1];
        const float m = a > c ? a : c;
        const float ea = expf(a - m), ec = expf(c - m);
        out[b * 2 + 0] = ea / (ea + ec);
        out[b * 2 + 1] = ec / (ea + ec);
    }
}

// ---------------- host driver ------------------------------------------------
extern "C" void kernel_launch(void* const* d_in, const int* in_sizes, int n_in,
                              void* d_out, int out_size, void* d_ws, size_t ws_size,
                              hipStream_t stream) {
    (void)in_sizes; (void)n_in; (void)out_size; (void)ws_size;
    const int*   word1 = (const int*)d_in[0];
    const int*   word2 = (const int*)d_in[1];
    const int*   piny1 = (const int*)d_in[2];
    const int*   piny2 = (const int*)d_in[3];
    const int*   len1  = (const int*)d_in[4];
    const int*   len2  = (const int*)d_in[5];
    const int*   tb1   = (const int*)d_in[6];
    const int*   te1   = (const int*)d_in[7];
    const int*   tb2   = (const int*)d_in[8];
    const int*   te2   = (const int*)d_in[9];
    const float* embw  = (const float*)d_in[10];
    const float* embp  = (const float*)d_in[11];
    const float* pe    = (const float*)d_in[12];
    const float* wih   = (const float*)d_in[13];   // [2][1536][832]
    const float* whh   = (const float*)d_in[14];   // [2][1536][512]
    const float* bih   = (const float*)d_in[15];   // [2][1536]
    const float* bhh   = (const float*)d_in[16];   // [2][1536]
    const float* cw0   = (const float*)d_in[17];
    const float* cw1   = (const float*)d_in[18];
    const float* cw2   = (const float*)d_in[19];
    const float* cbias = (const float*)d_in[20];
    const float* bng   = (const float*)d_in[21];
    const float* bnb   = (const float*)d_in[22];
    const float* bnm   = (const float*)d_in[23];
    const float* bnv   = (const float*)d_in[24];
    const float* fcw   = (const float*)d_in[25];
    const float* fcb   = (const float*)d_in[26];

    char* ws = (char*)d_ws;
    size_t off = 0;
    auto alloc = [&](size_t bytes) -> void* {
        void* p = ws + off;
        off += (bytes + 255) & ~(size_t)255;
        return p;
    };

    __bf16* x_bf0   = (__bf16*)alloc((size_t)kNBL * kDIN * 2);
    __bf16* x_bf1   = (__bf16*)alloc((size_t)kNBL * kDIN * 2);
    __bf16* wih_bf  = (__bf16*)alloc((size_t)2 * k3H * kDIN * 2);
    __bf16* whh_bf  = (__bf16*)alloc((size_t)2 * k3H * kH * 2);
    float*  gi      = (float*)alloc((size_t)4 * kNBL * k3H * 4);
    float*  gout    = (float*)alloc((size_t)4 * kNBL * kH * 4);
    __bf16* enc_bf0 = (__bf16*)alloc((size_t)kNBL * kENC * 2);
    __bf16* enc_bf1 = (__bf16*)alloc((size_t)kNBL * kENC * 2);
    float*  norms0  = (float*)alloc((size_t)kNBL * 4);
    float*  norms1  = (float*)alloc((size_t)kNBL * 4);
    float*  sim     = (float*)alloc((size_t)kB * kL * kL * 4);
    float*  maxs    = (float*)alloc((size_t)2 * kB * kL * 4);
    float*  rel     = (float*)alloc((size_t)2 * kB * kL * 4);
    float*  feat    = (float*)alloc((size_t)kB * 768 * 4);

    // 1) convert GRU weights to bf16
    { int n = 2 * k3H * kDIN; f32_to_bf16_kernel<<<(n + 255) / 256, 256, 0, stream>>>(wih, wih_bf, n); }
    { int n = 2 * k3H * kH;   f32_to_bf16_kernel<<<(n + 255) / 256, 256, 0, stream>>>(whh, whh_bf, n); }

    // 2) embedding concat per side
    embed_kernel<<<dim3(kNBL), 256, 0, stream>>>(word1, piny1, tb1, te1, embw, embp, pe, x_bf0);
    embed_kernel<<<dim3(kNBL), 256, 0, stream>>>(word2, piny2, tb2, te2, embw, embp, pe, x_bf1);

    // 3) gi = x @ Wih^T + bih  for (side, dir) -- WMMA GEMMs
    for (int side = 0; side < 2; ++side) {
        const __bf16* xb = side ? x_bf1 : x_bf0;
        for (int dir = 0; dir < 2; ++dir) {
            gemm_abt_bias_kernel<<<dim3(k3H / 64, kNBL / 128), 256, 0, stream>>>(
                xb, wih_bf + (size_t)dir * k3H * kDIN, bih + dir * k3H,
                gi + (size_t)(side * 2 + dir) * kNBL * k3H, kNBL, k3H, kDIN);
        }
    }

    // 4) persistent GRU scan (4 units), 288KB dynamic LDS per block
    {
        const int smem_bytes = (kB * kH + kB * k3H) * (int)sizeof(float) + kB * kH * 2; // 294912
        hipFuncSetAttribute(reinterpret_cast<const void*>(gru_scan_kernel),
                            hipFuncAttributeMaxDynamicSharedMemorySize, smem_bytes);
        gru_scan_kernel<<<4, 1024, smem_bytes, stream>>>(gi, whh_bf, bhh, len1, len2, gout);
    }

    // 5) build bf16 encodings + norms
    build_enc_kernel<<<kNBL, 256, 0, stream>>>(gout + (size_t)0 * kNBL * kH,
                                               gout + (size_t)1 * kNBL * kH, len1, enc_bf0, norms0);
    build_enc_kernel<<<kNBL, 256, 0, stream>>>(gout + (size_t)2 * kNBL * kH,
                                               gout + (size_t)3 * kNBL * kH, len2, enc_bf1, norms1);

    // 6) batched cosine similarity (WMMA + TDM-staged B panel)
    cossim_kernel<<<dim3(kL / 64, kL / 128, kB), 256, 0, stream>>>(enc_bf0, enc_bf1, norms0, norms1, sim);

    // 7) max/argmax over axis 2 (axis=0 here) and axis 1 (axis=1 here)
    maxarg_kernel<<<dim3(2, kB), kL, 0, stream>>>(sim, maxs, rel);

    // 8) conv branches + BN + relu + maxpool
    conv_branch_kernel<<<2 * kB, kCONV, 0, stream>>>(maxs, rel, cw0, cw1, cw2, cbias,
                                                     bng, bnb, bnm, bnv, feat);

    // 9) fc + softmax
    fc_softmax_kernel<<<kB, 64, 0, stream>>>(feat, fcw, fcb, (float*)d_out);
}
